// Token_performer_55731495632924
// MI455X (gfx1250) — matrix operand order repaired and checked by reading, not compiled, not run on hardware
//
#include <hip/hip_runtime.h>
#include <hip/hip_bf16.h>
#include <math.h>

// ---------------- problem constants ----------------
constexpr int Bsz = 64;        // batch
constexpr int Tn  = 3136;      // tokens
constexpr int IND = 147;       // input dim
constexpr int Dm  = 64;        // model dim
constexpr int Mrf = 32;        // random features
constexpr int KP  = 160;       // 147 padded to multiple of 32
constexpr int TT  = 64;        // tokens per workgroup tile
constexpr int NT  = Tn / TT;   // 49 tiles per batch
constexpr float EPSF    = 1e-8f;
constexpr float RSQRT_M = 0.17677669529663687f;   // 1/sqrt(32)

// ---------------- WMMA fragment helpers (gfx1250, wave32) ----------------
typedef __attribute__((ext_vector_type(16))) _Float16 v16h;
typedef __attribute__((ext_vector_type(8)))  float    v8f;

union Frag16 { v16h v; _Float16 h[16]; uint4 q[2]; };

// A 16x32 f16: lanes 0-15 -> M=lane, K = 0..7 then 16..23 ; lanes 16-31 -> K = 8..15 then 24..31
__device__ __forceinline__ v16h load_a(const _Float16* A, int ld, int m0, int k0, int lane) {
  Frag16 f;
  int m = lane & 15, g = lane >> 4;
  const _Float16* p = A + (m0 + m) * ld + k0 + g * 8;
  f.q[0] = *(const uint4*)(p);
  f.q[1] = *(const uint4*)(p + 16);
  return f.v;
}
// B 32x16 f16 from transposed storage Bt[n][k]: lane n=l&15, K = (l>=16? 8:0) + 0..7 then +16..23
__device__ __forceinline__ v16h load_bt(const _Float16* Bt, int ld, int n0, int k0, int lane) {
  Frag16 f;
  int n = lane & 15, g = lane >> 4;
  const _Float16* p = Bt + (n0 + n) * ld + k0 + g * 8;
  f.q[0] = *(const uint4*)(p);
  f.q[1] = *(const uint4*)(p + 16);
  return f.v;
}
// A fragment where the logical A is the transpose of LDS storage S (S[k][m], ld = row stride of S)
__device__ __forceinline__ v16h load_a_tr(const _Float16* S, int ld, int m0, int k0, int lane) {
  Frag16 f;
  int m = lane & 15, g = lane >> 4;
#pragma unroll
  for (int j = 0; j < 8; ++j) {
    f.h[j]     = S[(k0 + g * 8 + j) * ld      + m0 + m];
    f.h[j + 8] = S[(k0 + 16 + g * 8 + j) * ld + m0 + m];
  }
  return f.v;
}
__device__ __forceinline__ v8f wmma_f16(v16h a, v16h b, v8f c) {
  return __builtin_amdgcn_wmma_f32_16x16x32_f16(false, a, false, b, (short)0, c, false, false);
}

// Async DMA: global -> LDS, 16 bytes per lane, tracked on ASYNCcnt (gfx1250 TDM-lite path)
__device__ __forceinline__ void async_copy_b128(unsigned lds_byte_off, const void* gptr) {
  unsigned long long ga = (unsigned long long)(size_t)gptr;
  asm volatile("global_load_async_to_lds_b128 %0, %1, off"
               :: "v"(lds_byte_off), "v"(ga) : "memory");
}
__device__ __forceinline__ void wait_async0() {
  asm volatile("s_wait_asynccnt 0x0" ::: "memory");
}

// =================================================================
// Pass 1: x -> proj1 -> LN1 -> kqv -> FAVOR+ features, partial ks / k^T v
// one workgroup = one batch-tile of 64 tokens, 256 threads (8 waves)
// =================================================================
__global__ __launch_bounds__(256) void tp_pass1(
    const float* __restrict__ x,    const float* __restrict__ W1,   const float* __restrict__ b1,
    const float* __restrict__ Wkqv, const float* __restrict__ bkqv,
    const float* __restrict__ g1,   const float* __restrict__ be1,
    const float* __restrict__ wrf,
    float* __restrict__ qp_ws, float* __restrict__ v_ws,
    float* __restrict__ kptv_ws, float* __restrict__ ks_ws)
{
  __shared__ alignas(16) char smem[61440];
  // Region 1 (24576 B): x tile -> Wkqv^T -> kqv (f16)
  _Float16* xs    = (_Float16*)smem;            // [64][160]
  _Float16* wkqvt = (_Float16*)smem;            // [192][64]
  _Float16* kqv16 = (_Float16*)smem;            // [64][192]
  // Region 2 (20480 B): W1^T -> { hn16 | wrf^T | kp^T | stats | xd }
  char* R2 = smem + 24576;
  _Float16* w1t   = (_Float16*)R2;              // [64][160]
  _Float16* hn16  = (_Float16*)R2;              // [64][64]   (8192 B)
  _Float16* wrft  = (_Float16*)(R2 + 8192);     // [32][64]   (4096 B)
  _Float16* kpT   = (_Float16*)(R2 + 12288);    // [32][64]   (4096 B) kp transposed [m][t]
  float*    stats = (float*)(R2 + 16384);       // [2][64] sum/sumsq
  float*    xd    = (float*)(R2 + 16896);       // [2][64] 0.5*|k|^2 , 0.5*|q|^2
  // Region 3 (16384 B)
  char* R3 = smem + 45056;
  _Float16* v16s  = (_Float16*)R3;              // [64][64] v tile f16
  float*    kp32  = (float*)(R3 + 8192);        // [64][32] kp f32

  const int tid  = threadIdx.x;
  const int lane = tid & 31;
  const int wv   = tid >> 5;                    // 0..7
  const int bb   = blockIdx.x / NT;
  const int t0   = (blockIdx.x % NT) * TT;
  const size_t tokbase = (size_t)bb * Tn + t0;

  // ---- P0: stage x tile and W1^T (K padded to 160) ----
  for (int i = tid; i < TT * KP; i += 256) {
    int r = i / KP, k = i % KP;
    xs[i] = (k < IND) ? (_Float16)x[(tokbase + r) * IND + k] : (_Float16)0.f;
  }
  for (int i = tid; i < Dm * KP; i += 256) {
    int n = i / KP, k = i % KP;
    w1t[i] = (k < IND) ? (_Float16)W1[(size_t)k * Dm + n] : (_Float16)0.f;
  }
  __syncthreads();

  // ---- P1: h = x @ W1 + b1  (64x64 out, 16 tiles, 2 per wave, K=160) ----
  float hreg[2][8];
  for (int i = 0; i < 2; ++i) {
    int id = wv * 2 + i, mi = id >> 2, ni = id & 3;
    v8f c = {};
#pragma unroll
    for (int ks = 0; ks < KP / 32; ++ks)
      c = wmma_f16(load_a(xs, KP, mi * 16, ks * 32, lane),
                   load_bt(w1t, KP, ni * 16, ks * 32, lane), c);
    float bias = b1[ni * 16 + (lane & 15)];
#pragma unroll
    for (int r = 0; r < 8; ++r) hreg[i][r] = c[r] + bias;
  }
  __syncthreads();

  // ---- P2a: zero stats+xd; stage Wkqv^T and wrf^T ----
  ((float*)(R2 + 16384))[tid] = 0.f;            // 256 floats: stats + xd
  for (int i = tid; i < 192 * Dm; i += 256) {
    int n = i / Dm, k = i % Dm;
    wkqvt[i] = (_Float16)Wkqv[(size_t)k * 192 + n];
  }
  for (int i = tid; i < Mrf * Dm; i += 256) wrft[i] = (_Float16)wrf[i];
  __syncthreads();

  // ---- P2b: LN1 stats via LDS atomics on fragments ----
  for (int i = 0; i < 2; ++i) {
    int id = wv * 2 + i, mi = id >> 2;
#pragma unroll
    for (int r = 0; r < 8; ++r) {
      int row = mi * 16 + r + (lane >> 4) * 8;
      float v = hreg[i][r];
      atomicAdd(&stats[row], v);
      atomicAdd(&stats[64 + row], v * v);
    }
  }
  __syncthreads();

  // ---- P2c: normalize -> hn16 ----
  for (int i = 0; i < 2; ++i) {
    int id = wv * 2 + i, mi = id >> 2, ni = id & 3;
    int col = ni * 16 + (lane & 15);
    float gg = g1[col], bbv = be1[col];
#pragma unroll
    for (int r = 0; r < 8; ++r) {
      int row = mi * 16 + r + (lane >> 4) * 8;
      float mu  = stats[row] * (1.f / 64.f);
      float var = stats[64 + row] * (1.f / 64.f) - mu * mu;
      hn16[row * Dm + col] = (_Float16)((hreg[i][r] - mu) * rsqrtf(var + 1e-5f) * gg + bbv);
    }
  }
  __syncthreads();

  // ---- P3: kqv = hn @ Wkqv + b (64x192, 48 tiles, 6 per wave, K=64) ----
  float kqvreg[6][8];
  for (int i = 0; i < 6; ++i) {
    int id = wv * 6 + i, mi = id & 3, ni = id >> 2;      // ni 0..11
    v8f c = {};
    c = wmma_f16(load_a(hn16, Dm, mi * 16, 0, lane),  load_bt(wkqvt, Dm, ni * 16, 0, lane),  c);
    c = wmma_f16(load_a(hn16, Dm, mi * 16, 32, lane), load_bt(wkqvt, Dm, ni * 16, 32, lane), c);
    float bias = bkqv[ni * 16 + (lane & 15)];
#pragma unroll
    for (int r = 0; r < 8; ++r) kqvreg[i][r] = c[r] + bias;
  }
  __syncthreads();   // wkqvt reads done; R1 now reusable as kqv16

  // ---- P3b: stash kqv f16, v tile (LDS + global), 0.5|k|^2 / 0.5|q|^2 ----
  for (int i = 0; i < 6; ++i) {
    int id = wv * 6 + i, mi = id & 3, ni = id >> 2;
    int col = ni * 16 + (lane & 15);
#pragma unroll
    for (int r = 0; r < 8; ++r) {
      int row = mi * 16 + r + (lane >> 4) * 8;
      float v = kqvreg[i][r];
      kqv16[row * 192 + col] = (_Float16)v;
      if (ni < 4) {
        atomicAdd(&xd[row], 0.5f * v * v);
      } else if (ni < 8) {
        atomicAdd(&xd[64 + row], 0.5f * v * v);
      } else {
        v16s[row * Dm + (col - 128)] = (_Float16)v;
        v_ws[(tokbase + row) * Dm + (col - 128)] = v;
      }
    }
  }
  __syncthreads();

  // ---- P4: FAVOR+ features: {k,q} @ wrf^T -> exp(. - xd)/sqrt(M) ----
  for (int i = 0; i < 2; ++i) {
    int id = wv * 2 + i;
    int sel = id >> 3;                    // 0 = k, 1 = q (uniform per wave)
    int sub = id & 7, mi = sub >> 1, ni = sub & 1;
    v8f c = {};
    c = wmma_f16(load_a(kqv16 + sel * 64, 192, mi * 16, 0, lane),  load_bt(wrft, Dm, ni * 16, 0, lane),  c);
    c = wmma_f16(load_a(kqv16 + sel * 64, 192, mi * 16, 32, lane), load_bt(wrft, Dm, ni * 16, 32, lane), c);
    int col = ni * 16 + (lane & 15);
#pragma unroll
    for (int r = 0; r < 8; ++r) {
      int row = mi * 16 + r + (lane >> 4) * 8;
      float val = expf(c[r] - xd[sel * 64 + row]) * RSQRT_M;
      if (sel == 0) {
        kp32[row * Mrf + col] = val;
        kpT[col * Dm + row]   = (_Float16)val;   // transposed for B-frag
      } else {
        qp_ws[(tokbase + row) * Mrf + col] = val;
      }
    }
  }
  __syncthreads();

  // ---- P5: partial ks and kptv (A = v^T, K = tokens) ----
  if (tid < Mrf) {
    float s = 0.f;
    for (int t = 0; t < TT; ++t) s += kp32[t * Mrf + tid];
    atomicAdd(&ks_ws[bb * Mrf + tid], s);
  }
  {
    int mi = wv >> 1, ni = wv & 1;                 // 8 tiles, 1 per wave
    v8f c = {};
    c = wmma_f16(load_a_tr(v16s, Dm, mi * 16, 0, lane),  load_bt(kpT, Dm, ni * 16, 0, lane),  c);
    c = wmma_f16(load_a_tr(v16s, Dm, mi * 16, 32, lane), load_bt(kpT, Dm, ni * 16, 32, lane), c);
#pragma unroll
    for (int r = 0; r < 8; ++r) {
      int d = mi * 16 + r + (lane >> 4) * 8;
      int m = ni * 16 + (lane & 15);
      atomicAdd(&kptv_ws[((size_t)bb * Dm + d) * Mrf + m], c[r]);
    }
  }
}

// =================================================================
// Pass 2: attn out / proj2 + residual / LN2 / GELU-MLP / residual
// v-tile and qp-tile are contiguous -> async DMA straight into LDS
// =================================================================
__global__ __launch_bounds__(256) void tp_pass2(
    const float* __restrict__ qp_ws, const float* __restrict__ v_ws,
    const float* __restrict__ kptv_ws, const float* __restrict__ ks_ws,
    const float* __restrict__ Wp2, const float* __restrict__ bp2,
    const float* __restrict__ g2,  const float* __restrict__ be2,
    const float* __restrict__ Wm1, const float* __restrict__ bm1,
    const float* __restrict__ Wm2, const float* __restrict__ bm2,
    float* __restrict__ out)
{
  __shared__ alignas(16) char smem[58368];
  float*    sY     = (float*)smem;                        // [64][64] v -> y (f32), async target
  char* R2 = smem + 16384;
  _Float16* qp16   = (_Float16*)R2;                       // [64][32]
  _Float16* kptv16 = (_Float16*)(R2 + 4096);              // [64][32] Bt[d][m]
  float*    qpraw  = (float*)(R2 + 8192);                 // [64][32] f32, async target
  _Float16* ya16   = (_Float16*)(R2 + 8192);              // [64][64] reuse after qpraw consumed
  _Float16* yn16   = (_Float16*)R2;                       // reuse after attn
  _Float16* gl16   = (_Float16*)(R2 + 8192);              // reuse after proj2
  _Float16* wp2t   = (_Float16*)(smem + 32768);           // [64][64]
  _Float16* wm1t   = (_Float16*)(smem + 40960);
  _Float16* wm2t   = (_Float16*)(smem + 49152);
  float*    sD     = (float*)(smem + 57344);              // [64]
  float*    sKS    = (float*)(smem + 57600);              // [32]
  float*    sG2    = (float*)(smem + 57728);              // [64]
  float*    sBe2   = (float*)(smem + 57984);              // [64]

  const int tid  = threadIdx.x;
  const int lane = tid & 31;
  const int wv   = tid >> 5;
  const int bb   = blockIdx.x / NT;
  const int t0   = (blockIdx.x % NT) * TT;
  const size_t tokbase = (size_t)bb * Tn + t0;

  // ---- P0: async DMA the contiguous tiles, regular loads for the rest ----
  {
    const char* vsrc = (const char*)(v_ws + tokbase * Dm);     // 16384 B contiguous
    for (int i = tid; i < 1024; i += 256)
      async_copy_b128((unsigned)(i * 16), vsrc + (size_t)i * 16);
    const char* qsrc = (const char*)(qp_ws + tokbase * Mrf);   // 8192 B contiguous
    for (int i = tid; i < 512; i += 256)
      async_copy_b128((unsigned)(16384 + 8192 + i * 16), qsrc + (size_t)i * 16);
  }
  for (int i = tid; i < Dm * Mrf; i += 256)
    kptv16[i] = (_Float16)kptv_ws[(size_t)bb * Dm * Mrf + i];
  for (int i = tid; i < Dm * Dm; i += 256) {
    int n = i >> 6, k = i & 63;
    wp2t[i] = (_Float16)Wp2[k * Dm + n];
    wm1t[i] = (_Float16)Wm1[k * Dm + n];
    wm2t[i] = (_Float16)Wm2[k * Dm + n];
  }
  if (tid < Mrf) sKS[tid] = ks_ws[bb * Mrf + tid];
  if (tid < Dm)  { sG2[tid] = g2[tid]; sBe2[tid] = be2[tid]; }
  wait_async0();          // my wave's async LDS writes are done
  __syncthreads();        // everyone's are

  // ---- P0b: qp f32 -> f16, and D = qp . ks ----
  for (int i = tid; i < TT * Mrf; i += 256) qp16[i] = (_Float16)qpraw[i];
  if (tid < TT) {
    float s = 0.f;
    for (int m = 0; m < Mrf; ++m) s += qpraw[tid * Mrf + m] * sKS[m];
    sD[tid] = s;
  }
  __syncthreads();

  // ---- P1: attn = (qp @ kptv^T) / (D + eps)  (K=32, 1 step) ----
  for (int i = 0; i < 2; ++i) {
    int id = wv * 2 + i, mi = id >> 2, ni = id & 3;
    v8f c = {};
    c = wmma_f16(load_a(qp16, Mrf, mi * 16, 0, lane), load_bt(kptv16, Mrf, ni * 16, 0, lane), c);
    int col = ni * 16 + (lane & 15);
#pragma unroll
    for (int r = 0; r < 8; ++r) {
      int row = mi * 16 + r + (lane >> 4) * 8;
      ya16[row * Dm + col] = (_Float16)(c[r] / (sD[row] + EPSF));
    }
  }
  __syncthreads();

  // ---- P2: y = v + attn @ Wp2 + bp2 (in-place in sY) ----
  for (int i = 0; i < 2; ++i) {
    int id = wv * 2 + i, mi = id >> 2, ni = id & 3;
    v8f c = {};
    c = wmma_f16(load_a(ya16, Dm, mi * 16, 0, lane),  load_bt(wp2t, Dm, ni * 16, 0, lane),  c);
    c = wmma_f16(load_a(ya16, Dm, mi * 16, 32, lane), load_bt(wp2t, Dm, ni * 16, 32, lane), c);
    int col = ni * 16 + (lane & 15);
    float bias = bp2[col];
#pragma unroll
    for (int r = 0; r < 8; ++r) {
      int row = mi * 16 + r + (lane >> 4) * 8;
      sY[row * Dm + col] += c[r] + bias;
    }
  }
  __syncthreads();

  // ---- P3: LN2 -> yn16 (skewed column order: bank-conflict-free row scans) ----
  if (tid < TT) {
    float mu = 0.f;
    for (int j = 0; j < Dm; ++j) mu += sY[tid * Dm + ((j + tid) & 63)];
    mu *= (1.f / 64.f);
    float var = 0.f;
    for (int j = 0; j < Dm; ++j) {
      float d = sY[tid * Dm + ((j + tid) & 63)] - mu;
      var += d * d;
    }
    float rs = rsqrtf(var * (1.f / 64.f) + 1e-5f);
    for (int j = 0; j < Dm; ++j) {
      int jj = (j + tid) & 63;
      yn16[tid * Dm + jj] = (_Float16)((sY[tid * Dm + jj] - mu) * rs * sG2[jj] + sBe2[jj]);
    }
  }
  __syncthreads();

  // ---- P4: gelu(yn @ Wm1 + bm1) (exact erf) ----
  for (int i = 0; i < 2; ++i) {
    int id = wv * 2 + i, mi = id >> 2, ni = id & 3;
    v8f c = {};
    c = wmma_f16(load_a(yn16, Dm, mi * 16, 0, lane),  load_bt(wm1t, Dm, ni * 16, 0, lane),  c);
    c = wmma_f16(load_a(yn16, Dm, mi * 16, 32, lane), load_bt(wm1t, Dm, ni * 16, 32, lane), c);
    int col = ni * 16 + (lane & 15);
    float bias = bm1[col];
#pragma unroll
    for (int r = 0; r < 8; ++r) {
      int row = mi * 16 + r + (lane >> 4) * 8;
      float t = c[r] + bias;
      gl16[row * Dm + col] = (_Float16)(0.5f * t * (1.f + erff(t * 0.70710678118654752f)));
    }
  }
  __syncthreads();

  // ---- P5: out = y + g @ Wm2 + bm2 ----
  for (int i = 0; i < 2; ++i) {
    int id = wv * 2 + i, mi = id >> 2, ni = id & 3;
    v8f c = {};
    c = wmma_f16(load_a(gl16, Dm, mi * 16, 0, lane),  load_bt(wm2t, Dm, ni * 16, 0, lane),  c);
    c = wmma_f16(load_a(gl16, Dm, mi * 16, 32, lane), load_bt(wm2t, Dm, ni * 16, 32, lane), c);
    int col = ni * 16 + (lane & 15);
    float bias = bm2[col];
#pragma unroll
    for (int r = 0; r < 8; ++r) {
      int row = mi * 16 + r + (lane >> 4) * 8;
      out[(tokbase + row) * Dm + col] = sY[row * Dm + col] + c[r] + bias;
    }
  }
}

// =================================================================
extern "C" void kernel_launch(void* const* d_in, const int* in_sizes, int n_in,
                              void* d_out, int out_size, void* d_ws, size_t ws_size,
                              hipStream_t stream) {
  (void)in_sizes; (void)n_in; (void)out_size; (void)ws_size;
  const float* x    = (const float*)d_in[0];
  const float* W1   = (const float*)d_in[1];
  const float* b1   = (const float*)d_in[2];
  const float* Wkqv = (const float*)d_in[3];
  const float* bkqv = (const float*)d_in[4];
  const float* Wp2  = (const float*)d_in[5];
  const float* bp2  = (const float*)d_in[6];
  const float* g1   = (const float*)d_in[7];
  const float* be1  = (const float*)d_in[8];
  const float* g2   = (const float*)d_in[9];
  const float* be2  = (const float*)d_in[10];
  const float* Wm1  = (const float*)d_in[11];
  const float* bm1  = (const float*)d_in[12];
  const float* Wm2  = (const float*)d_in[13];
  const float* bm2  = (const float*)d_in[14];
  const float* wrf  = (const float*)d_in[15];
  float* out = (float*)d_out;

  char* ws = (char*)d_ws;
  size_t off = 0;
  float* v_ws    = (float*)(ws + off); off += (size_t)Bsz * Tn * Dm  * 4;
  float* qp_ws   = (float*)(ws + off); off += (size_t)Bsz * Tn * Mrf * 4;
  float* kptv_ws = (float*)(ws + off); off += (size_t)Bsz * Dm * Mrf * 4;
  float* ks_ws   = (float*)(ws + off); off += (size_t)Bsz * Mrf * 4;

  // zero the cross-tile accumulators (kptv + ks are contiguous)
  hipMemsetAsync(kptv_ws, 0, (size_t)Bsz * Dm * Mrf * 4 + (size_t)Bsz * Mrf * 4, stream);

  dim3 grid(Bsz * NT), blk(256);
  tp_pass1<<<grid, blk, 0, stream>>>(x, W1, b1, Wkqv, bkqv, g1, be1, wrf,
                                     qp_ws, v_ws, kptv_ws, ks_ws);
  tp_pass2<<<grid, blk, 0, stream>>>(qp_ws, v_ws, kptv_ws, ks_ws,
                                     Wp2, bp2, g2, be2, Wm1, bm1, Wm2, bm2, out);
}
